// MultiHeadAttention_25331717112501
// MI455X (gfx1250) — matrix operand order repaired and checked by reading, not compiled, not run on hardware
//
#include <hip/hip_runtime.h>
#include <hip/hip_bf16.h>

// MHA forward: B=2,S=2048,E=768,H=8,D=96 (fp32 in/out), bf16 WMMA compute.
// Attention streams K/V tiles to LDS via the Tensor Data Mover (TDM).

typedef __bf16 bf16_t;
typedef __attribute__((ext_vector_type(16))) __bf16 v16bf;
typedef __attribute__((ext_vector_type(8)))  __bf16 v8bf;
typedef __attribute__((ext_vector_type(8)))  float  v8f;
typedef __attribute__((ext_vector_type(4)))  unsigned int u32x4;
typedef __attribute__((ext_vector_type(8)))  int i32x8;
typedef __attribute__((ext_vector_type(4)))  int i32x4;

#define NB   2
#define NS   2048
#define NE   768
#define NH   8
#define ND   96
#define SCALING 0.10206207261596575f   // 96^-0.5

#if defined(__has_builtin)
#  if __has_builtin(__builtin_amdgcn_tensor_load_to_lds) && \
      __has_builtin(__builtin_amdgcn_s_wait_tensorcnt)
#    define HAVE_TDM 1
#  endif
#endif
#ifndef HAVE_TDM
#  define HAVE_TDM 0
#endif

static __device__ inline v16bf cat16(v8bf lo, v8bf hi) {
  v16bf r;
#pragma unroll
  for (int i = 0; i < 8; ++i) { r[i] = lo[i]; r[i + 8] = hi[i]; }
  return r;
}

// 16 contiguous bf16 (32B, 16B-aligned) -> v16bf
static __device__ inline v16bf ld16(const bf16_t* p) {
  v8bf lo = *(const v8bf*)(p);
  v8bf hi = *(const v8bf*)(p + 8);
  return cat16(lo, hi);
}

static __device__ inline v8f wmma_bf16(v16bf a, v16bf b, v8f c) {
  return __builtin_amdgcn_wmma_f32_16x16x32_bf16(
      /*neg_a=*/false, a, /*neg_b=*/false, b,
      /*c_mod=*/(short)0, c, /*reuse_a=*/false, /*reuse_b=*/false);
}

#if HAVE_TDM
// Issue a TDM 2-D tile load (bf16 elements) into LDS.
// Descriptor layout per CDNA5 ISA (D# groups 0/1).
static __device__ inline void tdm_load_2d(unsigned lds_byte_addr,
                                          const void* gptr,
                                          unsigned tile_d0, unsigned tile_d1,
                                          unsigned tensor_d0, unsigned tensor_d1,
                                          unsigned long long stride0_elems) {
  unsigned long long ga = (unsigned long long)(uintptr_t)gptr;
  u32x4 g0;
  g0[0] = 1u;                                         // count=1 valid, user mode
  g0[1] = lds_byte_addr;                              // lds_addr [63:32]
  g0[2] = (unsigned)(ga & 0xffffffffu);               // global_addr[31:0]
  g0[3] = (unsigned)((ga >> 32) & 0x01ffffffu)        // global_addr[56:32]
          | (2u << 30);                               // type=2 ("image")
  i32x8 g1;
  g1[0] = (int)(1u << 16);                            // wg_mask=0, data_size=1 (2B)
  g1[1] = (int)((tensor_d0 & 0xffffu) << 16);         // tensor_dim0[15:0]
  g1[2] = (int)(((tensor_d0 >> 16) & 0xffffu) |
                ((tensor_d1 & 0xffffu) << 16));       // dim0 hi | dim1 lo
  g1[3] = (int)(((tensor_d1 >> 16) & 0xffffu) |
                ((tile_d0 & 0xffffu) << 16));         // dim1 hi | tile_dim0
  g1[4] = (int)(tile_d1 & 0xffffu);                   // tile_dim1, tile_dim2=0
  g1[5] = (int)(unsigned)(stride0_elems & 0xffffffffu);
  g1[6] = (int)(unsigned)((stride0_elems >> 32) & 0xffffu);  // stride0 hi, stride1=0
  g1[7] = 0;
  i32x4 z4 = {0, 0, 0, 0};
#if __clang_major__ >= 23
  i32x8 z8 = {0, 0, 0, 0, 0, 0, 0, 0};
  __builtin_amdgcn_tensor_load_to_lds(g0, g1, z4, z4, z8, 0);
#else
  __builtin_amdgcn_tensor_load_to_lds(g0, g1, z4, z4, 0);
#endif
}
#endif  // HAVE_TDM

// ---------------------------------------------------------------------------
// Projection GEMM: out = X[M,768] @ W[768,768] + bias, bf16 into [B,H,S,D]
// (optionally also [B,H,D,S]). Block tile 64x64, 8 waves = 4(m) x 2(n), each
// wave 16x32 (A-fragment reused across two WMMAs).
// ---------------------------------------------------------------------------
__global__ __launch_bounds__(256) void proj_kernel(
    const float* __restrict__ X,     // [M][768] fp32
    const float* __restrict__ W,     // [768][768] fp32
    const float* __restrict__ bias,  // [768]
    bf16_t* __restrict__ out_bhsd,   // [B,H,S,D] bf16
    bf16_t* __restrict__ out_bhds,   // [B,H,D,S] bf16 or nullptr
    int M) {
  __shared__ __align__(16) __bf16 As[64][32];   // [m][k]
  __shared__ __align__(16) __bf16 Bs[64][32];   // [n][k]

  const int tid  = threadIdx.x;
  const int lane = tid & 31;
  const int wave = tid >> 5;
  const int mT   = wave & 3;   // 0..3
  const int nT   = wave >> 2;  // 0..1
  const int mBlocks = M / 64;
  const int m0 = (blockIdx.x % mBlocks) * 64;
  const int n0 = (blockIdx.x / mBlocks) * 64;

  const int mrow  = mT * 16 + (lane & 15);
  const int ncol0 = nT * 32 + (lane & 15);
  const int ksel  = lane >> 4;

  v8f acc[2] = {};
  for (int k0 = 0; k0 < NE; k0 += 32) {
#pragma unroll 4
    for (int i = tid; i < 64 * 32; i += 256) {   // stage A (coalesced over k)
      int kl = i & 31, ml = i >> 5;
      As[ml][kl] = (__bf16)X[(size_t)(m0 + ml) * NE + k0 + kl];
    }
#pragma unroll 4
    for (int i = tid; i < 64 * 32; i += 256) {   // stage B^T [n][k] (coalesced n)
      int nl = i & 63, kl = i >> 6;
      Bs[nl][kl] = (__bf16)W[(size_t)(k0 + kl) * NE + n0 + nl];
    }
    __syncthreads();
    v16bf a = ld16(&As[mrow][16 * ksel]);
    v16bf b0 = ld16(&Bs[ncol0][16 * ksel]);
    v16bf b1 = ld16(&Bs[ncol0 + 16][16 * ksel]);
    acc[0] = wmma_bf16(a, b0, acc[0]);
    acc[1] = wmma_bf16(a, b1, acc[1]);
    __syncthreads();
  }

#pragma unroll
  for (int t = 0; t < 2; ++t) {
    const int nG = n0 + nT * 32 + t * 16 + (lane & 15);
    const float bv = bias[nG];
    const int h = nG / ND, d = nG % ND;
#pragma unroll
    for (int r = 0; r < 8; ++r) {
      int mG = m0 + mT * 16 + r + 8 * ksel;   // C layout row = r + 8*(lane>>4)
      int bb = mG >> 11, ss = mG & (NS - 1);
      float v = acc[t][r] + bv;
      out_bhsd[(((size_t)(bb * NH + h)) * NS + ss) * ND + d] = (__bf16)v;
      if (out_bhds)
        out_bhds[(((size_t)(bb * NH + h)) * ND + d) * NS + ss] = (__bf16)v;
    }
  }
}

// ---------------------------------------------------------------------------
// Flash attention: 8 waves/block, all on the same (b,h); each wave owns a
// 16-query tile. K/V chunks (32 keys) staged once per block into LDS by the
// Tensor Data Mover, consumed by all 8 waves.
// ---------------------------------------------------------------------------
__global__ __launch_bounds__(256) void attn_kernel(
    const bf16_t* __restrict__ Q,    // [B*H, S, 96]
    const bf16_t* __restrict__ K,    // [B*H, S, 96]   (also V)
    const bf16_t* __restrict__ Kt,   // [B*H, 96, S]   (V transposed)
    bf16_t* __restrict__ O) {        // [B*S, 768] attention output, bf16
  __shared__ __align__(16) __bf16 Ks[32][96];     // keys x d   (6 KB)
  __shared__ __align__(16) __bf16 Kts[96][32];    // d x keys   (6 KB)
  __shared__ __align__(16) __bf16 Pl[8][16][32];  // per-wave P bounce (8 KB)

  const int tid  = threadIdx.x;
  const int lane = tid & 31;
  const int wave = tid >> 5;
  const int qt = blockIdx.x * 8 + wave;     // 0..2047; 8 tiles/block, same bh
  const int bh = qt >> 7;
  const int q0 = (qt & 127) * 16;

  const bf16_t* Qp  = Q  + (size_t)bh * NS * ND;
  const bf16_t* Kp  = K  + (size_t)bh * NS * ND;
  const bf16_t* Ktp = Kt + (size_t)bh * ND * NS;

  const int m    = lane & 15;
  const int ksel = lane >> 4;

  // Q fragments: 3 chunks of 16x32 over d (from global, wave-private rows)
  v16bf qf[3];
#pragma unroll
  for (int c = 0; c < 3; ++c) {
    const bf16_t* base = Qp + (size_t)(q0 + m) * ND + c * 32;
    qf[c] = ld16(base + 16 * ksel);
  }

  float rowmax[8], rowsum[8];
#pragma unroll
  for (int r = 0; r < 8; ++r) { rowmax[r] = -3.0e30f; rowsum[r] = 0.0f; }
  v8f oacc[6] = {};

  for (int j0 = 0; j0 < NS; j0 += 32) {
    __syncthreads();   // previous chunk's consumers done before overwrite
#if HAVE_TDM
    if (wave == 0) {
      // K chunk: 2-D tile 96(d, contig) x 32(keys), row stride 96
      tdm_load_2d((unsigned)(uintptr_t)(void*)&Ks[0][0],
                  Kp + (size_t)j0 * ND, ND, 32, ND, NS, ND);
      // Kt chunk: 2-D tile 32(keys, contig) x 96(d), row stride 2048
      tdm_load_2d((unsigned)(uintptr_t)(void*)&Kts[0][0],
                  Ktp + j0, 32, ND, NS, ND, NS);
      __builtin_amdgcn_s_wait_tensorcnt(0);
    }
#else
    for (int i = tid; i < 32 * ND; i += 256) {
      int dc = i % ND, kk = i / ND;
      Ks[kk][dc] = Kp[(size_t)(j0 + kk) * ND + dc];
    }
    for (int i = tid; i < ND * 32; i += 256) {
      int kk = i & 31, dd = i >> 5;
      Kts[dd][kk] = Ktp[(size_t)dd * NS + j0 + kk];
    }
#endif
    __syncthreads();

    // ---- scores: two 16x16 tiles (keys j0..15 / j0+16..31), K-depth 96 ----
    v8f s0 = {}, s1 = {};
#pragma unroll
    for (int c = 0; c < 3; ++c) {
      v16bf b0 = ld16(&Ks[m][c * 32 + 16 * ksel]);
      v16bf b1 = ld16(&Ks[m + 16][c * 32 + 16 * ksel]);
      s0 = wmma_bf16(qf[c], b0, s0);
      s1 = wmma_bf16(qf[c], b1, s1);
    }

    // ---- online softmax over this 32-key chunk ----
#pragma unroll
    for (int r = 0; r < 8; ++r) {
      float mx = fmaxf(s0[r], s1[r]);
      mx = fmaxf(mx, __shfl_xor(mx, 1, 32));
      mx = fmaxf(mx, __shfl_xor(mx, 2, 32));
      mx = fmaxf(mx, __shfl_xor(mx, 4, 32));
      mx = fmaxf(mx, __shfl_xor(mx, 8, 32));
      float nm = fmaxf(rowmax[r], mx);
      float sc = __expf(rowmax[r] - nm);
      rowmax[r] = nm;
      float p0 = __expf(s0[r] - nm);
      float p1 = __expf(s1[r] - nm);
      float ps = p0 + p1;
      ps += __shfl_xor(ps, 1, 32);
      ps += __shfl_xor(ps, 2, 32);
      ps += __shfl_xor(ps, 4, 32);
      ps += __shfl_xor(ps, 8, 32);
      rowsum[r] = rowsum[r] * sc + ps;
      int mr = r + 8 * ksel;                // C-layout -> LDS bounce
      Pl[wave][mr][m] = (__bf16)p0;
      Pl[wave][mr][m + 16] = (__bf16)p1;
#pragma unroll
      for (int c = 0; c < 6; ++c) oacc[c][r] *= sc;
    }
    // same-wave LDS RAW: DS ops are in-order; force completion + no reorder
    asm volatile("s_wait_dscnt 0x0" ::: "memory");

    // reload P as a 16x32 A-fragment
    v16bf pf = ld16(&Pl[wave][m][16 * ksel]);

    // ---- out += P @ V : 6 d-chunks of 16 columns ----
#pragma unroll
    for (int c = 0; c < 6; ++c) {
      v16bf bv = ld16(&Kts[c * 16 + m][16 * ksel]);
      oacc[c] = wmma_bf16(pf, bv, oacc[c]);
    }
  }

  // ---- epilogue: (softmax / rowsum) * SCALING, store [B*S, 768] bf16 ----
  const int bb = bh >> 3, hh = bh & 7;
#pragma unroll
  for (int r = 0; r < 8; ++r) {
    float inv = SCALING / rowsum[r];
    int srow = q0 + r + 8 * ksel;
    size_t base = ((size_t)bb * NS + srow) * NE + hh * ND;
#pragma unroll
    for (int c = 0; c < 6; ++c)
      O[base + c * 16 + m] = (__bf16)(oacc[c][r] * inv);
  }
}

// ---------------------------------------------------------------------------
// Output projection: d_out = A[4096,768](bf16) @ Wo[768,768] + bo, fp32 out.
// Same 64x64 block tile / 16x32 per-wave shape as proj_kernel.
// ---------------------------------------------------------------------------
__global__ __launch_bounds__(256) void oproj_kernel(
    const bf16_t* __restrict__ A,    // [M][768] bf16
    const float* __restrict__ W,     // [768][768] fp32
    const float* __restrict__ bias,  // [768]
    float* __restrict__ out,         // [M][768] fp32
    int M) {
  __shared__ __align__(16) __bf16 As[64][32];
  __shared__ __align__(16) __bf16 Bs[64][32];

  const int tid  = threadIdx.x;
  const int lane = tid & 31;
  const int wave = tid >> 5;
  const int mT   = wave & 3;
  const int nT   = wave >> 2;
  const int mBlocks = M / 64;
  const int m0 = (blockIdx.x % mBlocks) * 64;
  const int n0 = (blockIdx.x / mBlocks) * 64;

  const int mrow  = mT * 16 + (lane & 15);
  const int ncol0 = nT * 32 + (lane & 15);
  const int ksel  = lane >> 4;

  v8f acc[2] = {};
  for (int k0 = 0; k0 < NE; k0 += 32) {
#pragma unroll 4
    for (int i = tid; i < 64 * 32; i += 256) {
      int kl = i & 31, ml = i >> 5;
      As[ml][kl] = A[(size_t)(m0 + ml) * NE + k0 + kl];
    }
#pragma unroll 4
    for (int i = tid; i < 64 * 32; i += 256) {
      int nl = i & 63, kl = i >> 6;
      Bs[nl][kl] = (__bf16)W[(size_t)(k0 + kl) * NE + n0 + nl];
    }
    __syncthreads();
    v16bf a = ld16(&As[mrow][16 * ksel]);
    v16bf b0 = ld16(&Bs[ncol0][16 * ksel]);
    v16bf b1 = ld16(&Bs[ncol0 + 16][16 * ksel]);
    acc[0] = wmma_bf16(a, b0, acc[0]);
    acc[1] = wmma_bf16(a, b1, acc[1]);
    __syncthreads();
  }

#pragma unroll
  for (int t = 0; t < 2; ++t) {
    const int nG = n0 + nT * 32 + t * 16 + (lane & 15);
    const float bv = bias[nG];
#pragma unroll
    for (int r = 0; r < 8; ++r) {
      int mG = m0 + mT * 16 + r + 8 * ksel;
      out[(size_t)mG * NE + nG] = acc[t][r] + bv;
    }
  }
}

// ---------------------------------------------------------------------------
extern "C" void kernel_launch(void* const* d_in, const int* in_sizes, int n_in,
                              void* d_out, int out_size, void* d_ws, size_t ws_size,
                              hipStream_t stream) {
  (void)in_sizes; (void)n_in; (void)out_size; (void)ws_size;
  const float* x  = (const float*)d_in[0];
  const float* Wq = (const float*)d_in[1];
  const float* bq = (const float*)d_in[2];
  const float* Wk = (const float*)d_in[3];
  const float* bk = (const float*)d_in[4];
  const float* Wo = (const float*)d_in[5];
  const float* bo = (const float*)d_in[6];
  float* out = (float*)d_out;

  const size_t NQK = (size_t)NB * NH * NS * ND;  // 3,145,728 elems
  bf16_t* Qw  = (bf16_t*)d_ws;
  bf16_t* Kw  = Qw  + NQK;
  bf16_t* Ktw = Kw  + NQK;
  bf16_t* Ow  = Ktw + NQK;   // [B*S, E] bf16

  const int M = NB * NS;                         // 4096
  const int projGrid = (M / 64) * (NE / 64);     // 64 * 12 = 768
  dim3 blk(256);

  proj_kernel<<<projGrid, blk, 0, stream>>>(x, Wq, bq, Qw, nullptr, M);
  proj_kernel<<<projGrid, blk, 0, stream>>>(x, Wk, bk, Kw, Ktw, M);
  attn_kernel<<<(NB * NH * NS / 16) / 8, blk, 0, stream>>>(Qw, Kw, Ktw, Ow);
  oproj_kernel<<<projGrid, blk, 0, stream>>>(Ow, Wo, bo, out, M);
}
